// BasicBlockDiscriminator_60627758350826
// MI455X (gfx1250) — matrix operand order repaired and checked by reading, not compiled, not run on hardware
//
#include <hip/hip_runtime.h>
#include <hip/hip_bf16.h>

#define NB      2
#define NNODE   65536
#define FIN     64
#define FOUT    128
#define NEDGE   393216
#define NPOOL   4
#define NNEXT   16384          // NNODE / NPOOL
#define NROWS   (NB * NNODE)   // 131072 flattened (b,n) rows

typedef __attribute__((ext_vector_type(16))) __bf16 v16bf;
typedef __attribute__((ext_vector_type(8)))  float  v8f;

// ---------------------------------------------------------------------------
// LDS weight layout: fragment-major ("swizzled") so one lane's B fragment is
// a contiguous 32-byte run:
//   sWf[ ((f*8 + ct)*32 + lane)*16 + j ]  ==  W[32f + 16*(lane/16) + j][16ct + lane%16]
// Lane fetch = single 32B vector load (2x ds_load_b128), zero packing VALU.
// ---------------------------------------------------------------------------
__device__ inline void stage_weights_swizzled(const float* __restrict__ W,
                                              __bf16* __restrict__ sWf,
                                              int K /* 64 or 128 */) {
    for (int i = threadIdx.x; i < K * FOUT; i += 256) {
        const int k  = i / FOUT;
        const int n  = i % FOUT;
        const int f  = k >> 5;            // K-fragment
        const int kk = k & 31;
        const int hh = (kk >> 4) & 1;     // wave half
        const int j  = kk & 15;           // element within fragment
        const int ct = n >> 4;            // column tile
        const int r  = n & 15;            // lane within half
        sWf[(((f * 8 + ct) * 32) + (hh * 16 + r)) * 16 + j] = (__bf16)W[i];
    }
}

__device__ inline v16bf load_bfrag(const __bf16* __restrict__ sWf,
                                   int f, int ct, int lane) {
    return *reinterpret_cast<const v16bf*>(
        sWf + (((f * 8 + ct) * 32) + lane) * 16);
}

// ---------------------------------------------------------------------------
// Wave-level GEMM helper: accumulate one 16x16 output column-tile.
// Emits NF x v_wmma_f32_16x16x32_bf16, each fed by one 32B LDS fragment load.
// ---------------------------------------------------------------------------
template <int NF>
__device__ inline v8f wmma_coltile(const v16bf* a, const __bf16* sWf,
                                   int ct, int lane) {
    v8f acc = {};
#pragma unroll
    for (int f = 0; f < NF; ++f) {
        const v16bf bfr = load_bfrag(sWf, f, ct, lane);
        acc = __builtin_amdgcn_wmma_f32_16x16x32_bf16(
                  false, a[f], false, bfr, (short)0, acc, false, false);
    }
    return acc;
}

// Build one bf16 A-fragment (K window [32*f, 32*f+32)) from an f32 LDS row.
// The 16 elements are one contiguous 16-float run (k = 16h + s) -> 4x float4 LDS loads.
template <bool LRELU>
__device__ inline v16bf make_afrag(const float* Arow, int f, int h) {
    const float4 q0 = *(const float4*)(Arow + f * 32 + 16 * h);
    const float4 q1 = *(const float4*)(Arow + f * 32 + 16 * h + 4);
    const float4 q2 = *(const float4*)(Arow + f * 32 + 16 * h + 8);
    const float4 q3 = *(const float4*)(Arow + f * 32 + 16 * h + 12);
    const float t[16] = {q0.x, q0.y, q0.z, q0.w, q1.x, q1.y, q1.z, q1.w,
                         q2.x, q2.y, q2.z, q2.w, q3.x, q3.y, q3.z, q3.w};
    v16bf a;
#pragma unroll
    for (int j = 0; j < 16; ++j) {
        float v = t[j];
        if (LRELU) v = (v > 0.0f) ? v : 0.2f * v;
        a[j] = (__bf16)v;
    }
    return a;
}

// ---------------------------------------------------------------------------
// Degree / inverse degree
// ---------------------------------------------------------------------------
__global__ void k_deg(const int* __restrict__ edge_start,
                      float* __restrict__ deg) {
    int e = blockIdx.x * blockDim.x + threadIdx.x;
    if (e < NEDGE) atomicAdd(&deg[edge_start[e]], 1.0f);
}

__global__ void k_invdeg(const float* __restrict__ deg,
                         float* __restrict__ invd) {
    int n = blockIdx.x * blockDim.x + threadIdx.x;
    if (n < NNODE) invd[n] = 1.0f / fmaxf(deg[n], 1.0f);
}

// ---------------------------------------------------------------------------
// Stage 1: residual + ECC0 linears.
// Block = 256 threads (8 waves), 128 rows per block.
// LDS: 128x64 f32 A-tile (32 KB) + 3 x swizzled 64x128 bf16 weights (48 KB).
// ---------------------------------------------------------------------------
__global__ void k_lin0(const float* __restrict__ x,
                       const float* __restrict__ W_nbr0,
                       const float* __restrict__ W_self0,
                       const float* __restrict__ b_self0,
                       const float* __restrict__ W_res,
                       const float* __restrict__ b_res,
                       float* __restrict__ h0,
                       float* __restrict__ self0,
                       float* __restrict__ rpool) {
    extern __shared__ __align__(32) char smem[];
    float*  sA = (float*)smem;                       // 128 x 64 f32
    __bf16* sW = (__bf16*)(smem + 128 * FIN * 4);    // 3 x swizzled 64x128

    const int rowbase = blockIdx.x * 128;            // flat (b*N + n) row

    const float4* xsrc = (const float4*)(x + (size_t)rowbase * FIN);
    float4* adst = (float4*)sA;
    for (int i = threadIdx.x; i < 128 * FIN / 4; i += 256) adst[i] = xsrc[i];
    stage_weights_swizzled(W_nbr0,  sW,                  FIN);
    stage_weights_swizzled(W_self0, sW + FIN * FOUT,     FIN);
    stage_weights_swizzled(W_res,   sW + 2 * FIN * FOUT, FIN);
    __syncthreads();

    const int wv   = threadIdx.x >> 5;
    const int lane = threadIdx.x & 31;
    const int h    = lane >> 4;
    const int r    = lane & 15;
    const float* Arow = sA + (wv * 16 + r) * FIN;

    v16bf aL[2], aR[2];                              // lrelu'd and raw A frags
#pragma unroll
    for (int f = 0; f < 2; ++f) {
        aL[f] = make_afrag<true >(Arow, f, h);
        aR[f] = make_afrag<false>(Arow, f, h);
    }

    const int wrow = rowbase + wv * 16;              // first row of this wave
#pragma unroll
    for (int ct = 0; ct < 8; ++ct) {
        v8f accH = wmma_coltile<2>(aL, sW,                  ct, lane);
        v8f accS = wmma_coltile<2>(aL, sW + FIN * FOUT,     ct, lane);
        v8f accR = wmma_coltile<2>(aR, sW + 2 * FIN * FOUT, ct, lane);
        const int col = ct * 16 + r;
        const float bs = b_self0[col];
        const float br = b_res[col];
#pragma unroll
        for (int v = 0; v < 8; ++v) {
            const int row = wrow + v + 8 * h;        // = b*N + n
            const size_t o = (size_t)row * FOUT + col;
            h0[o]    = accH[v];
            self0[o] = accS[v] + bs;
            // pooled residual: coarse row = b*NNEXT + (n >> 2)
            const int b = row >> 16;                 // N == 65536
            const int n = row & (NNODE - 1);
            atomicAdd(&rpool[((size_t)b * NNEXT + (n >> 2)) * FOUT + col],
                      0.25f * (accR[v] + br));
        }
    }
}

// ---------------------------------------------------------------------------
// Edge scatter: one wave per edge, both batches. L2-resident working set.
// ---------------------------------------------------------------------------
__global__ void k_scatter(const float* __restrict__ hbuf,
                          const int*   __restrict__ edge_start,
                          const int*   __restrict__ edge_end,
                          const float* __restrict__ adj,
                          float* __restrict__ agg) {
    const int e = (blockIdx.x * blockDim.x + threadIdx.x) >> 5;
    const int lane = threadIdx.x & 31;
    if (e >= NEDGE) return;
    const int   s = edge_start[e];
    const int   t = edge_end[e];
    const float a = adj[e];
#pragma unroll
    for (int b = 0; b < NB; ++b) {
        const float4 v =
            ((const float4*)(hbuf + ((size_t)b * NNODE + t) * FOUT))[lane];
        float* dst = agg + ((size_t)b * NNODE + s) * FOUT + lane * 4;
        atomicAdd(dst + 0, a * v.x);
        atomicAdd(dst + 1, a * v.y);
        atomicAdd(dst + 2, a * v.z);
        atomicAdd(dst + 3, a * v.w);
    }
}

// ---------------------------------------------------------------------------
// Stage 2: out1 = lrelu(self0 + inv_deg*agg0); h1 = out1@W_nbr1,
// self1 = out1@W_self1 + b_self1. K = 128 (4 WMMA K-fragments).
// LDS: 128x128 f32 A (64 KB) + 2 x swizzled 128x128 bf16 weights (64 KB).
// self1 overwrites self0 in place (block-disjoint rows); h1 reuses h0 buffer.
// ---------------------------------------------------------------------------
__global__ void k_lin1(const float* __restrict__ self0,
                       const float* __restrict__ agg0,
                       const float* __restrict__ invd,
                       const float* __restrict__ W_nbr1,
                       const float* __restrict__ W_self1,
                       const float* __restrict__ b_self1,
                       float* __restrict__ h1,
                       float* __restrict__ self1) {
    extern __shared__ __align__(32) char smem[];
    float*  sA = (float*)smem;                        // 128 x 128 f32
    __bf16* sW = (__bf16*)(smem + 128 * FOUT * 4);    // 2 x swizzled 128x128

    const int rowbase = blockIdx.x * 128;

    for (int i = threadIdx.x; i < 128 * FOUT; i += 256) {
        const int row = rowbase + i / FOUT;
        const int n   = row & (NNODE - 1);
        const size_t o = (size_t)rowbase * FOUT + i;
        const float v = self0[o] + invd[n] * agg0[o];
        sA[i] = (v > 0.0f) ? v : 0.2f * v;            // out1 (pre-GEMM act)
    }
    stage_weights_swizzled(W_nbr1,  sW,               FOUT);
    stage_weights_swizzled(W_self1, sW + FOUT * FOUT, FOUT);
    __syncthreads();

    const int wv   = threadIdx.x >> 5;
    const int lane = threadIdx.x & 31;
    const int h    = lane >> 4;
    const int r    = lane & 15;
    const float* Arow = sA + (wv * 16 + r) * FOUT;

    v16bf a[4];
#pragma unroll
    for (int f = 0; f < 4; ++f) a[f] = make_afrag<false>(Arow, f, h);

    const int wrow = rowbase + wv * 16;
#pragma unroll
    for (int ct = 0; ct < 8; ++ct) {
        v8f accH = wmma_coltile<4>(a, sW,               ct, lane);
        v8f accS = wmma_coltile<4>(a, sW + FOUT * FOUT, ct, lane);
        const int col = ct * 16 + r;
        const float bs = b_self1[col];
#pragma unroll
        for (int v = 0; v < 8; ++v) {
            const size_t o = (size_t)(wrow + v + 8 * h) * FOUT + col;
            h1[o]    = accH[v];
            self1[o] = accS[v] + bs;
        }
    }
}

// ---------------------------------------------------------------------------
// Final: out2 = self1 + inv_deg*agg1 ; pool by 4 ; add pooled residual.
// ---------------------------------------------------------------------------
__global__ void k_final(const float* __restrict__ self1,
                        const float* __restrict__ agg1,
                        const float* __restrict__ invd,
                        const float* __restrict__ rpool,
                        float* __restrict__ out) {
    const size_t i = (size_t)blockIdx.x * blockDim.x + threadIdx.x;
    if (i >= (size_t)NB * NNEXT * FOUT) return;
    const int    col  = (int)(i % FOUT);
    const size_t crow = i / FOUT;
    const int    b    = (int)(crow / NNEXT);
    const int    cn   = (int)(crow % NNEXT);
    const size_t fbase = ((size_t)b * NNODE + (size_t)cn * NPOOL) * FOUT + col;
    float s = 0.0f;
#pragma unroll
    for (int k = 0; k < NPOOL; ++k) {
        const size_t o = fbase + (size_t)k * FOUT;
        const int    n = cn * NPOOL + k;
        s += self1[o] + invd[n] * agg1[o];
    }
    out[i] = 0.25f * s + rpool[i];
}

// ---------------------------------------------------------------------------
// Launch
// ---------------------------------------------------------------------------
extern "C" void kernel_launch(void* const* d_in, const int* in_sizes, int n_in,
                              void* d_out, int out_size, void* d_ws,
                              size_t ws_size, hipStream_t stream) {
    const float* x       = (const float*)d_in[0];
    const float* W_res   = (const float*)d_in[1];
    const float* b_res   = (const float*)d_in[2];
    const float* W_self0 = (const float*)d_in[3];
    const float* b_self0 = (const float*)d_in[4];
    const float* W_nbr0  = (const float*)d_in[5];
    const float* W_self1 = (const float*)d_in[6];
    const float* b_self1 = (const float*)d_in[7];
    const float* W_nbr1  = (const float*)d_in[8];
    const float* adj     = (const float*)d_in[9];
    const int*   e_start = (const int*)d_in[11];
    const int*   e_end   = (const int*)d_in[12];
    float*       out     = (float*)d_out;

    // workspace carve-up (f32 elements)
    const size_t BIG = (size_t)NROWS * FOUT;          // 16,777,216
    float* h_buf = (float*)d_ws;                      // h0 then h1
    float* s_buf = h_buf + BIG;                       // self0 then self1
    float* agg   = s_buf + BIG;                       // agg0 then agg1
    float* rpool = agg + BIG;                         // pooled residual
    float* deg   = rpool + (size_t)NB * NNEXT * FOUT;
    float* invd  = deg + NNODE;

    hipMemsetAsync(deg,   0, NNODE * sizeof(float), stream);
    hipMemsetAsync(rpool, 0, (size_t)NB * NNEXT * FOUT * sizeof(float), stream);
    hipMemsetAsync(agg,   0, BIG * sizeof(float), stream);

    k_deg   <<<NEDGE / 256, 256, 0, stream>>>(e_start, deg);
    k_invdeg<<<NNODE / 256, 256, 0, stream>>>(deg, invd);

    const size_t lds0 = 128 * FIN * sizeof(float) + 3 * FIN * FOUT * sizeof(__bf16);
    k_lin0<<<NROWS / 128, 256, lds0, stream>>>(x, W_nbr0, W_self0, b_self0,
                                               W_res, b_res, h_buf, s_buf,
                                               rpool);

    k_scatter<<<NEDGE / 8, 256, 0, stream>>>(h_buf, e_start, e_end, adj, agg);

    const size_t lds1 = 128 * FOUT * sizeof(float) + 2 * FOUT * FOUT * sizeof(__bf16);
    k_lin1<<<NROWS / 128, 256, lds1, stream>>>(s_buf, agg, invd, W_nbr1,
                                               W_self1, b_self1, h_buf, s_buf);

    hipMemsetAsync(agg, 0, BIG * sizeof(float), stream);
    k_scatter<<<NEDGE / 8, 256, 0, stream>>>(h_buf, e_start, e_end, adj, agg);

    k_final<<<((size_t)NB * NNEXT * FOUT + 255) / 256, 256, 0, stream>>>(
        s_buf, agg, invd, rpool, out);
}
